// DualGNN_24713241821995
// MI455X (gfx1250) — matrix operand order repaired and checked by reading, not compiled, not run on hardware
//
#include <hip/hip_runtime.h>

typedef __attribute__((ext_vector_type(16))) _Float16 v16h;
typedef __attribute__((ext_vector_type(8)))  _Float16 v8h;
typedef __attribute__((ext_vector_type(8)))  float    v8f;

#define WMMA_F16(a, b, c) \
  __builtin_amdgcn_wmma_f32_16x16x32_f16(false, (a), false, (b), (short)0, (c), false, false)

static __device__ __forceinline__ float selu_f(float x) {
  const float alpha = 1.6732632423543772f;
  const float scale = 1.0507009873554805f;
  return x > 0.0f ? scale * x : scale * alpha * (expf(x) - 1.0f);
}

// A fragment (16x32) from f16 row-major activation X[ld]; two contiguous 8-half runs per lane.
static __device__ __forceinline__ v16h load_afrag_h(const _Float16* X, int ldx, int m0, int k0, int lane) {
  int m = m0 + (lane & 15);
  int g = (lane >> 4) & 1;
  const _Float16* row = X + (size_t)m * ldx + k0 + 8 * g;
  v8h lo = *(const v8h*)(row);        // K = k0+8g .. +7
  v8h hi = *(const v8h*)(row + 16);   // K = k0+8g+16 .. +23
  v16h a;
#pragma unroll
  for (int i = 0; i < 8; ++i) { a[i] = lo[i]; a[8 + i] = hi[i]; }
  return a;
}

// B fragment from swizzled weights: per (kt,ct) tile, lane's 16 halves are contiguous (32B).
static __device__ __forceinline__ v16h load_bfrag_sw(const _Float16* Wsw, int nct, int kt, int ct, int lane) {
  return *(const v16h*)(Wsw + (((size_t)(kt * nct + ct)) * 32 + lane) * 16);
}

// ---- weight conversion: W[N,K] fp32 row-major -> fragment-swizzled f16 ----
// dst[idx]: idx = (((kt*nct)+ct)*32 + lane)*16 + i ;  k = kt*32 + (i&7) + 8*(lane>>4) + 16*(i>>3), n = ct*16 + (lane&15)
__global__ void k_convw_sw(const float* W, _Float16* Wsw, int N, int K) {
  int idx = blockIdx.x * blockDim.x + threadIdx.x;
  if (idx >= N * K) return;
  int i = idx & 15;
  int l = (idx >> 4) & 31;
  int tile = idx >> 9;
  int nct = N >> 4;
  int ct = tile % nct;
  int kt = tile / nct;
  int k = kt * 32 + (i & 7) + 8 * (l >> 4) + ((i >> 3) << 4);
  int n = ct * 16 + (l & 15);
  Wsw[idx] = (_Float16)W[(size_t)n * K + k];
}

// conv weight w[cout=128, cin=128, tap=3]; logical Wt[k=tap*128+cin][n=cout], K=384, N=128
__global__ void k_convw_conv_sw(const float* W, _Float16* Wsw) {
  int idx = blockIdx.x * blockDim.x + threadIdx.x;
  if (idx >= 384 * 128) return;
  int i = idx & 15;
  int l = (idx >> 4) & 31;
  int tile = idx >> 9;
  int ct = tile & 7;        // nct = 8
  int kt = tile >> 3;
  int k = kt * 32 + (i & 7) + 8 * (l >> 4) + ((i >> 3) << 4);
  int n = ct * 16 + (l & 15);
  int cin = k & 127, tap = k >> 7;
  Wsw[idx] = (_Float16)W[((size_t)n * 128 + cin) * 3 + tap];
}

// ---------------- gather + FM bi-interaction (f16 outputs) ----------------
__global__ void k_gather_user(const int* cate, const int* ids,
                              const float* gemb, const float* aemb, const float* oemb,
                              const float* uemb, _Float16* fmS, _Float16* seq, int M) {
  int idx = blockIdx.x * blockDim.x + threadIdx.x;
  if (idx >= M * 64) return;
  int row = idx >> 6, e = idx & 63;
  const int* c = cate + (size_t)row * 3;
  float g = gemb[(size_t)c[0] * 64 + e];
  float a = aemb[(size_t)c[1] * 64 + e];
  float o = oemb[(size_t)c[2] * 64 + e];
  float s  = g + a + o;
  float sq = g * g + a * a + o * o;
  fmS[(size_t)row * 128 + e]      = (_Float16)(0.5f * (s * s - sq));
  fmS[(size_t)row * 128 + 64 + e] = (_Float16)s;
  seq[(size_t)row * 128 + e] = (_Float16)uemb[(size_t)ids[row] * 64 + e];
}

__global__ void k_gather_poi(const int* cate, const int* lm, const int* fac,
                             const int* rat, const int* loc, const int* ids,
                             const float* cemb, const float* lemb, const float* femb,
                             const float* remb, const float* loemb, const float* pemb,
                             _Float16* fmS, _Float16* seq, int M) {
  int idx = blockIdx.x * blockDim.x + threadIdx.x;
  if (idx >= M * 64) return;
  int row = idx >> 6, e = idx & 63;
  float s = 0.0f, sq = 0.0f, v;
  const int* cc = cate + (size_t)row * 3;
#pragma unroll
  for (int j = 0; j < 3; ++j) { v = cemb[(size_t)cc[j] * 64 + e]; s += v; sq += v * v; }
  const int* ll = lm + (size_t)row * 2;
#pragma unroll
  for (int j = 0; j < 2; ++j) { v = lemb[(size_t)ll[j] * 64 + e]; s += v; sq += v * v; }
  const int* ff = fac + (size_t)row * 4;
#pragma unroll
  for (int j = 0; j < 4; ++j) { v = femb[(size_t)ff[j] * 64 + e]; s += v; sq += v * v; }
  v = remb[(size_t)rat[row] * 64 + e]; s += v; sq += v * v;
  v = loemb[(size_t)loc[row] * 64 + e]; s += v; sq += v * v;
  fmS[(size_t)row * 128 + e]      = (_Float16)(0.5f * (s * s - sq));
  fmS[(size_t)row * 128 + 64 + e] = (_Float16)s;
  seq[(size_t)row * 128 + e] = (_Float16)pemb[(size_t)ids[row] * 64 + e];
}

// ------- int = selu(fm@Wbi^T+bbi) + selu(s@Wsi^T+bsi) -> seq[:,64:128] (f16) -------
__global__ __launch_bounds__(128) void k_int(const _Float16* fmS,
    const _Float16* WbiSw, const _Float16* WsiSw,
    const float* bbi, const float* bsi, _Float16* seq) {
  int wave = threadIdx.x >> 5, lane = threadIdx.x & 31;
  int m0 = (blockIdx.x * 4 + wave) * 16;
  const v8f vz = {};
  v8f acc1[4], acc2[4];
#pragma unroll
  for (int c = 0; c < 4; ++c) { acc1[c] = vz; acc2[c] = vz; }
#pragma unroll
  for (int kt = 0; kt < 2; ++kt) {
    v16h a1 = load_afrag_h(fmS,      128, m0, kt * 32, lane);
    v16h a2 = load_afrag_h(fmS + 64, 128, m0, kt * 32, lane);
#pragma unroll
    for (int ct = 0; ct < 4; ++ct) {
      acc1[ct] = WMMA_F16(a1, load_bfrag_sw(WbiSw, 4, kt, ct, lane), acc1[ct]);
      acc2[ct] = WMMA_F16(a2, load_bfrag_sw(WsiSw, 4, kt, ct, lane), acc2[ct]);
    }
  }
  int nl = lane & 15, g = (lane >> 4) & 1;
#pragma unroll
  for (int ct = 0; ct < 4; ++ct) {
    int col = ct * 16 + nl;
    float b1v = bbi[col], b2v = bsi[col];
#pragma unroll
    for (int r = 0; r < 8; ++r) {
      int m = m0 + r + 8 * g;
      seq[(size_t)m * 128 + 64 + col] =
          (_Float16)(selu_f(acc1[ct][r] + b1v) + selu_f(acc2[ct][r] + b2v));
    }
  }
}

// ------- fused = selu(A[M,128] @ W[128,64] + bias) -> fp32 -------
__global__ __launch_bounds__(128) void k_fused(const _Float16* A,
    const _Float16* WSw, const float* bias, float* out) {
  int wave = threadIdx.x >> 5, lane = threadIdx.x & 31;
  int m0 = (blockIdx.x * 4 + wave) * 16;
  const v8f vz = {};
  v8f acc[4];
#pragma unroll
  for (int c = 0; c < 4; ++c) acc[c] = vz;
#pragma unroll
  for (int kt = 0; kt < 4; ++kt) {
    v16h a = load_afrag_h(A, 128, m0, kt * 32, lane);
#pragma unroll
    for (int ct = 0; ct < 4; ++ct)
      acc[ct] = WMMA_F16(a, load_bfrag_sw(WSw, 4, kt, ct, lane), acc[ct]);
  }
  int nl = lane & 15, g = (lane >> 4) & 1;
#pragma unroll
  for (int ct = 0; ct < 4; ++ct) {
    int col = ct * 16 + nl;
    float bv = bias[col];
#pragma unroll
    for (int r = 0; r < 8; ++r) {
      int m = m0 + r + 8 * g;
      out[(size_t)m * 64 + col] = selu_f(acc[ct][r] + bv);
    }
  }
}

// ------- Conv1d(128->128, k=3, SAME) as implicit GEMM (K=384), relu, f16 out -------
static __device__ __forceinline__ v16h load_afrag_conv_h(const _Float16* seq, int m0, int kt, int lane) {
  int m = m0 + (lane & 15);
  int b = m / 50, n = m % 50;
  int g = (lane >> 4) & 1;
  int tap  = kt >> 2;          // 32-wide K tile lies inside one tap
  int cin0 = (kt & 3) * 32;
  int nn = n + tap - 1;
  v16h a;
  if (nn >= 0 && nn < 50) {
    const _Float16* row = seq + ((size_t)b * 50 + nn) * 128 + cin0 + 8 * g;
    v8h lo = *(const v8h*)(row);
    v8h hi = *(const v8h*)(row + 16);
#pragma unroll
    for (int i = 0; i < 8; ++i) { a[i] = lo[i]; a[8 + i] = hi[i]; }
  } else {
#pragma unroll
    for (int i = 0; i < 16; ++i) a[i] = (_Float16)0.0f;
  }
  return a;
}

__global__ __launch_bounds__(128) void k_conv(const _Float16* seq, const _Float16* WSw,
                                              const float* bias, _Float16* out) {
  int wave = threadIdx.x >> 5, lane = threadIdx.x & 31;
  int m0 = (blockIdx.x * 4 + wave) * 16;
  const v8f vz = {};
  v8f acc[8];
#pragma unroll
  for (int c = 0; c < 8; ++c) acc[c] = vz;
#pragma unroll
  for (int kt = 0; kt < 12; ++kt) {
    v16h a = load_afrag_conv_h(seq, m0, kt, lane);
#pragma unroll
    for (int ct = 0; ct < 8; ++ct)
      acc[ct] = WMMA_F16(a, load_bfrag_sw(WSw, 8, kt, ct, lane), acc[ct]);
  }
  int nl = lane & 15, g = (lane >> 4) & 1;
#pragma unroll
  for (int ct = 0; ct < 8; ++ct) {
    int col = ct * 16 + nl;
    float bv = bias[col];
#pragma unroll
    for (int r = 0; r < 8; ++r) {
      int m = m0 + r + 8 * g;
      out[(size_t)m * 128 + col] = (_Float16)fmaxf(acc[ct][r] + bv, 0.0f);
    }
  }
}

// ---------------- mean over 50 neighbors (deterministic) ----------------
__global__ void k_mean(const float* fused, float* mean, int Btot) {
  int idx = blockIdx.x * blockDim.x + threadIdx.x;
  if (idx >= Btot * 64) return;
  int b = idx >> 6, e = idx & 63;
  float s = 0.0f;
  for (int n = 0; n < 50; ++n) s += fused[((size_t)b * 50 + n) * 64 + e];
  mean[idx] = s * (1.0f / 50.0f);
}

// ------- fused GRU: weights in LDS, h-state in LDS, one wave per 16 batch rows -------
__global__ __launch_bounds__(128) void k_gru(const _Float16* convY,
    const _Float16* WihSw, const float* bih,
    const _Float16* WhhSw, const float* bhh, float* hop1) {
  __shared__ __align__(16) _Float16 wlds[36864];  // Wih frags [24576] then Whh frags [12288]
  __shared__ float hbuf[4][16 * 64];
  int wave = threadIdx.x >> 5, lane = threadIdx.x & 31;
  int b0 = (blockIdx.x * 4 + wave) * 16;
  // cooperative stage of swizzled weights into LDS
  {
    const uint4* s0 = (const uint4*)WihSw;   // 3072 uint4
    const uint4* s1 = (const uint4*)WhhSw;   // 1536 uint4
    uint4* d0 = (uint4*)wlds;
    uint4* d1 = (uint4*)(wlds + 24576);
    for (int i = threadIdx.x; i < 3072; i += 128) d0[i] = s0[i];
    for (int i = threadIdx.x; i < 1536; i += 128) d1[i] = s1[i];
  }
  float* h = hbuf[wave];
  for (int i = lane; i < 16 * 64; i += 32) h[i] = 0.0f;
  __syncthreads();
  int nl = lane & 15, g = (lane >> 4) & 1;
  const v8f vz = {};
  for (int t = 0; t < 50; ++t) {
    v8f acc[12], accN[4];
#pragma unroll
    for (int c = 0; c < 12; ++c) acc[c] = vz;
#pragma unroll
    for (int c = 0; c < 4; ++c) accN[c] = vz;
    // gi = x_t @ Wih^T  (K = 128); x_t = convY[(b*50 + t), :] (f16)
    const _Float16* xrow = convY + ((size_t)(b0 + nl) * 50 + t) * 128;
    if (t + 1 < 50) __builtin_prefetch(convY + ((size_t)(b0 + nl) * 50 + t + 1) * 128, 0, 0);
#pragma unroll
    for (int kt = 0; kt < 4; ++kt) {
      const _Float16* rp = xrow + kt * 32 + 8 * g;
      v8h lo = *(const v8h*)(rp);
      v8h hi = *(const v8h*)(rp + 16);
      v16h a;
#pragma unroll
      for (int i = 0; i < 8; ++i) { a[i] = lo[i]; a[8 + i] = hi[i]; }
#pragma unroll
      for (int ct = 0; ct < 12; ++ct)
        acc[ct] = WMMA_F16(a, load_bfrag_sw(wlds, 12, kt, ct, lane), acc[ct]);
    }
    // gh = h @ Whh^T (K = 64): r,z merged into acc[0..7], n-gate contribution separate
#pragma unroll
    for (int kt = 0; kt < 2; ++kt) {
      v16h a;
#pragma unroll
      for (int i = 0; i < 16; ++i) {
        int k = kt * 32 + (i & 7) + 8 * g + ((i >> 3) << 4);
        a[i] = (_Float16)h[nl * 64 + k];
      }
#pragma unroll
      for (int ct = 0; ct < 8; ++ct)
        acc[ct] = WMMA_F16(a, load_bfrag_sw(wlds + 24576, 12, kt, ct, lane), acc[ct]);
#pragma unroll
      for (int ct = 0; ct < 4; ++ct)
        accN[ct] = WMMA_F16(a, load_bfrag_sw(wlds + 24576, 12, kt, 8 + ct, lane), accN[ct]);
    }
    // gates + state update (each (m,e) owned by exactly one lane slot)
#pragma unroll
    for (int et = 0; et < 4; ++et) {
      int e = et * 16 + nl;
      float br = bih[e] + bhh[e];
      float bz = bih[64 + e] + bhh[64 + e];
      float bni = bih[128 + e];
      float bnh = bhh[128 + e];
#pragma unroll
      for (int r = 0; r < 8; ++r) {
        int m = r + 8 * g;
        float rg = 1.0f / (1.0f + expf(-(acc[et][r] + br)));
        float zg = 1.0f / (1.0f + expf(-(acc[4 + et][r] + bz)));
        float ng = tanhf(acc[8 + et][r] + bni + rg * (accN[et][r] + bnh));
        float hold = h[m * 64 + e];
        h[m * 64 + e] = (1.0f - zg) * ng + zg * hold;
      }
    }
  }
#pragma unroll
  for (int et = 0; et < 4; ++et) {
    int e = et * 16 + nl;
#pragma unroll
    for (int r = 0; r < 8; ++r) {
      int m = r + 8 * g;
      hop1[(size_t)(b0 + m) * 64 + e] = h[m * 64 + e];
    }
  }
}

// ---------------- final: sum branches, fc dot, biases, sigmoid ----------------
__global__ void k_final(const float* uself, const float* uhop, const float* umean,
                        const float* pself, const float* phop, const float* pmean,
                        const float* fcw, const float* fcb,
                        const float* ubias, const float* pbias, const float* miu,
                        const int* user, const int* poi, float* out, int Btot) {
  int b = blockIdx.x * blockDim.x + threadIdx.x;
  if (b >= Btot) return;
  float acc = fcb[0] + ubias[user[b]] + pbias[poi[b]] + miu[0];
  size_t o = (size_t)b * 64;
  for (int e = 0; e < 64; ++e)
    acc += (uself[o + e] + uhop[o + e] + umean[o + e]) * fcw[e];
  for (int e = 0; e < 64; ++e)
    acc += (pself[o + e] + phop[o + e] + pmean[o + e]) * fcw[64 + e];
  out[b] = 1.0f / (1.0f + expf(-acc));
}

extern "C" void kernel_launch(void* const* d_in, const int* in_sizes, int n_in,
                              void* d_out, int out_size, void* d_ws, size_t ws_size,
                              hipStream_t stream) {
  (void)in_sizes; (void)n_in; (void)out_size; (void)ws_size;
  const int Bt = 2048, NN = 50;
  const int Mh = Bt * NN; // 102400

  enum {
    IN_USER = 0, IN_POI, IN_USC, IN_UOID, IN_UOC,
    IN_PSC, IN_PSL, IN_PSF, IN_PSR, IN_PSLoc,
    IN_POID, IN_POC, IN_POL, IN_POF, IN_POR, IN_POLoc,
    P_UEMB = 16, P_PEMB, P_UBIAS, P_PBIAS, P_MIU,
    P_GEMB, P_AEMB, P_OEMB, P_CEMB, P_LEMB, P_FEMB, P_REMB, P_LOEMB,
    P_PSBI_W, P_PSBI_B, P_PSSI_W, P_PSSI_B,
    P_PHBI_W, P_PHBI_B, P_PHSI_W, P_PHSI_B,
    P_USBI_W, P_USBI_B, P_USSI_W, P_USSI_B,
    P_UHBI_W, P_UHBI_B, P_UHSI_W, P_UHSI_B,
    P_PCS_W, P_PCS_B, P_PCH_W, P_PCH_B,
    P_UCS_W, P_UCS_B, P_UCH_W, P_UCH_B,
    P_PGK_W, P_PGK_B, P_UGK_W, P_UGK_B,
    P_PG_WIH, P_PG_WHH, P_PG_BIH, P_PG_BHH,
    P_UG_WIH, P_UG_WHH, P_UG_BIH, P_UG_BHH,
    P_FC_W, P_FC_B
  };
  auto fin = [&](int i) { return (const float*)d_in[i]; };
  auto iin = [&](int i) { return (const int*)d_in[i]; };

  char* ws = (char*)d_ws;
  size_t off = 0;
  auto alloc = [&](size_t bytes) -> void* {
    void* p = ws + off;
    off = (off + bytes + 255) & ~(size_t)255;
    return p;
  };
  // fragment-swizzled f16 weights
  _Float16* WbiUh = (_Float16*)alloc(64 * 64 * 2);
  _Float16* WsiUh = (_Float16*)alloc(64 * 64 * 2);
  _Float16* WbiUs = (_Float16*)alloc(64 * 64 * 2);
  _Float16* WsiUs = (_Float16*)alloc(64 * 64 * 2);
  _Float16* WbiPh = (_Float16*)alloc(64 * 64 * 2);
  _Float16* WsiPh = (_Float16*)alloc(64 * 64 * 2);
  _Float16* WbiPs = (_Float16*)alloc(64 * 64 * 2);
  _Float16* WsiPs = (_Float16*)alloc(64 * 64 * 2);
  _Float16* WcUh  = (_Float16*)alloc(128 * 64 * 2);
  _Float16* WcUs  = (_Float16*)alloc(128 * 64 * 2);
  _Float16* WcPh  = (_Float16*)alloc(128 * 64 * 2);
  _Float16* WcPs  = (_Float16*)alloc(128 * 64 * 2);
  _Float16* WcvU  = (_Float16*)alloc(384 * 128 * 2);
  _Float16* WcvP  = (_Float16*)alloc(384 * 128 * 2);
  _Float16* WihU  = (_Float16*)alloc(128 * 192 * 2);
  _Float16* WihP  = (_Float16*)alloc(128 * 192 * 2);
  _Float16* WhhU  = (_Float16*)alloc(64 * 192 * 2);
  _Float16* WhhP  = (_Float16*)alloc(64 * 192 * 2);
  // activations
  _Float16* fmS  = (_Float16*)alloc((size_t)Mh * 128 * 2);
  _Float16* seq  = (_Float16*)alloc((size_t)Mh * 128 * 2);
  float* fused = (float*)alloc((size_t)Mh * 64 * 4);
  float* meanU = (float*)alloc((size_t)Bt * 64 * 4);
  float* meanP = (float*)alloc((size_t)Bt * 64 * 4);
  float* hopU  = (float*)alloc((size_t)Bt * 64 * 4);
  float* hopP  = (float*)alloc((size_t)Bt * 64 * 4);
  _Float16* sFmS = (_Float16*)alloc((size_t)Bt * 128 * 2);
  _Float16* sSeq = (_Float16*)alloc((size_t)Bt * 128 * 2);
  float* uself = (float*)alloc((size_t)Bt * 64 * 4);
  float* pself = (float*)alloc((size_t)Bt * 64 * 4);
  _Float16* convY = fmS;  // safe alias: fmS is dead once k_int has run (stream is serial)

  auto cw = [&](int wi, _Float16* dst, int N, int K) {
    int tot = N * K;
    k_convw_sw<<<(tot + 255) / 256, 256, 0, stream>>>(fin(wi), dst, N, K);
  };
  cw(P_UHBI_W, WbiUh, 64, 64);  cw(P_UHSI_W, WsiUh, 64, 64);
  cw(P_USBI_W, WbiUs, 64, 64);  cw(P_USSI_W, WsiUs, 64, 64);
  cw(P_PHBI_W, WbiPh, 64, 64);  cw(P_PHSI_W, WsiPh, 64, 64);
  cw(P_PSBI_W, WbiPs, 64, 64);  cw(P_PSSI_W, WsiPs, 64, 64);
  cw(P_UCH_W, WcUh, 64, 128);   cw(P_UCS_W, WcUs, 64, 128);
  cw(P_PCH_W, WcPh, 64, 128);   cw(P_PCS_W, WcPs, 64, 128);
  k_convw_conv_sw<<<192, 256, 0, stream>>>(fin(P_UGK_W), WcvU);
  k_convw_conv_sw<<<192, 256, 0, stream>>>(fin(P_PGK_W), WcvP);
  cw(P_UG_WIH, WihU, 192, 128); cw(P_PG_WIH, WihP, 192, 128);
  cw(P_UG_WHH, WhhU, 192, 64);  cw(P_PG_WHH, WhhP, 192, 64);

  const int gemmHop = Mh / 64;   // 1600 blocks x 4 waves x 16 rows
  const int gemmSelf = Bt / 64;  // 32 blocks

  // ---------- USER branch ----------
  k_gather_user<<<(Mh * 64 + 255) / 256, 256, 0, stream>>>(
      iin(IN_UOC), iin(IN_UOID), fin(P_GEMB), fin(P_AEMB), fin(P_OEMB), fin(P_UEMB),
      fmS, seq, Mh);
  k_int<<<gemmHop, 128, 0, stream>>>(fmS, WbiUh, WsiUh, fin(P_UHBI_B), fin(P_UHSI_B), seq);
  k_fused<<<gemmHop, 128, 0, stream>>>(seq, WcUh, fin(P_UCH_B), fused);
  k_mean<<<(Bt * 64 + 255) / 256, 256, 0, stream>>>(fused, meanU, Bt);
  k_conv<<<gemmHop, 128, 0, stream>>>(seq, WcvU, fin(P_UGK_B), convY);
  k_gru<<<Bt / 64, 128, 0, stream>>>(convY, WihU, fin(P_UG_BIH), WhhU, fin(P_UG_BHH), hopU);
  // user self
  k_gather_user<<<(Bt * 64 + 255) / 256, 256, 0, stream>>>(
      iin(IN_USC), iin(IN_USER), fin(P_GEMB), fin(P_AEMB), fin(P_OEMB), fin(P_UEMB),
      sFmS, sSeq, Bt);
  k_int<<<gemmSelf, 128, 0, stream>>>(sFmS, WbiUs, WsiUs, fin(P_USBI_B), fin(P_USSI_B), sSeq);
  k_fused<<<gemmSelf, 128, 0, stream>>>(sSeq, WcUs, fin(P_UCS_B), uself);

  // ---------- POI branch ----------
  k_gather_poi<<<(Mh * 64 + 255) / 256, 256, 0, stream>>>(
      iin(IN_POC), iin(IN_POL), iin(IN_POF), iin(IN_POR), iin(IN_POLoc), iin(IN_POID),
      fin(P_CEMB), fin(P_LEMB), fin(P_FEMB), fin(P_REMB), fin(P_LOEMB), fin(P_PEMB),
      fmS, seq, Mh);
  k_int<<<gemmHop, 128, 0, stream>>>(fmS, WbiPh, WsiPh, fin(P_PHBI_B), fin(P_PHSI_B), seq);
  k_fused<<<gemmHop, 128, 0, stream>>>(seq, WcPh, fin(P_PCH_B), fused);
  k_mean<<<(Bt * 64 + 255) / 256, 256, 0, stream>>>(fused, meanP, Bt);
  k_conv<<<gemmHop, 128, 0, stream>>>(seq, WcvP, fin(P_PGK_B), convY);
  k_gru<<<Bt / 64, 128, 0, stream>>>(convY, WihP, fin(P_PG_BIH), WhhP, fin(P_PG_BHH), hopP);
  // poi self
  k_gather_poi<<<(Bt * 64 + 255) / 256, 256, 0, stream>>>(
      iin(IN_PSC), iin(IN_PSL), iin(IN_PSF), iin(IN_PSR), iin(IN_PSLoc), iin(IN_POI),
      fin(P_CEMB), fin(P_LEMB), fin(P_FEMB), fin(P_REMB), fin(P_LOEMB), fin(P_PEMB),
      sFmS, sSeq, Bt);
  k_int<<<gemmSelf, 128, 0, stream>>>(sFmS, WbiPs, WsiPs, fin(P_PSBI_B), fin(P_PSSI_B), sSeq);
  k_fused<<<gemmSelf, 128, 0, stream>>>(sSeq, WcPs, fin(P_PCS_B), pself);

  // ---------- final ----------
  k_final<<<(Bt + 255) / 256, 256, 0, stream>>>(
      uself, hopU, meanU, pself, hopP, meanP,
      fin(P_FC_W), fin(P_FC_B), fin(P_UBIAS), fin(P_PBIAS), fin(P_MIU),
      iin(IN_USER), iin(IN_POI), (float*)d_out, Bt);
}
